// AgentGraphPolicyNetwork_16655883174349
// MI455X (gfx1250) — matrix operand order repaired and checked by reading, not compiled, not run on hardware
//
#include <hip/hip_runtime.h>
#include <hip/hip_bf16.h>
#include <math.h>

#define N_NODES 50000
#define N_EDGES 800000

typedef __attribute__((ext_vector_type(16))) _Float16 v16h;
typedef __attribute__((ext_vector_type(8)))  _Float16 v8h;
typedef __attribute__((ext_vector_type(4)))  _Float16 v4h;
typedef __attribute__((ext_vector_type(8)))  float    v8f;

union AF { v16h v; v8h h[2]; };

// Branch-free tanh: native v_tanh_f32 on gfx1250 if available, else 1 - 2*rcp(exp(2x)+1)
__device__ __forceinline__ float fast_tanh(float x) {
#if __has_builtin(__builtin_amdgcn_tanhf)
  return __builtin_amdgcn_tanhf(x);
#else
  float e = __expf(2.f * x);                       // v_exp_f32 (TRANS)
  return 1.f - 2.f * __builtin_amdgcn_rcpf(e + 1.f); // saturates correctly at +/-1
#endif
}

// A fragment 16x32 f16 from LDS staging s[row][stride]:
// lanes 0-15: row = lane, K = {k0..k0+7, k0+16..k0+23}
// lanes 16-31: row = lane-16, K = {k0+8..k0+15, k0+24..k0+31}
__device__ __forceinline__ v16h ldsA(const _Float16* s, int stride, int k0, int lane) {
  int r = lane & 15, hf = lane >> 4;
  AF u;
  u.h[0] = *(const v8h*)(s + r * stride + k0 + hf * 8);
  u.h[1] = *(const v8h*)(s + r * stride + k0 + 16 + hf * 8);
  return u.v;
}
// B fragment 32x16 f16 from LDS weights w[out][K] (row-major):
// lanes 0-15: N = lane, K = {k0..k0+7, k0+16..k0+23} ; lanes 16-31: N = lane-16, K = {k0+8..k0+15, k0+24..k0+31}
__device__ __forceinline__ v16h ldsB(const _Float16* w, int K, int ntile, int k0, int lane) {
  int n = ntile * 16 + (lane & 15), hf = lane >> 4;
  const _Float16* p = w + n * K + k0 + hf * 8;
  AF u;
  u.h[0] = *(const v8h*)p;
  u.h[1] = *(const v8h*)(p + 16);
  return u.v;
}
__device__ __forceinline__ v8f wmma_f16(v16h a, v16h b, v8f c) {
  return __builtin_amdgcn_wmma_f32_16x16x32_f16(false, a, false, b, (short)0, c, false, false);
}

// 64->64 tanh layer, in-place through LDS staging (wave-private; LDS ops in-order per wave)
__device__ __forceinline__ void layer64_tanh(_Float16* s, const _Float16* w, const float* b, int lane) {
  v8f acc[4];
  int c0 = lane & 15;
#pragma unroll
  for (int t = 0; t < 4; t++) { float bv = b[t * 16 + c0];
#pragma unroll
    for (int j = 0; j < 8; j++) acc[t][j] = bv; }
  v16h a0 = ldsA(s, 64, 0, lane);
  v16h a1 = ldsA(s, 64, 32, lane);
#pragma unroll
  for (int t = 0; t < 4; t++) {
    acc[t] = wmma_f16(a0, ldsB(w, 64, t, 0, lane), acc[t]);
    acc[t] = wmma_f16(a1, ldsB(w, 64, t, 32, lane), acc[t]);
  }
  int rb = (lane >> 4) << 3;   // D layout: lanes 0-15 -> rows 0-7, lanes 16-31 -> rows 8-15
#pragma unroll
  for (int t = 0; t < 4; t++)
#pragma unroll
    for (int j = 0; j < 8; j++)
      s[(rb + j) * 64 + t * 16 + c0] = (_Float16)fast_tanh(acc[t][j]);
}

// -------- encoder: tanh MLP3 (in_dim -> 64 -> 64 -> 64), out f32 or f16 --------
__global__ void __launch_bounds__(256) mlp3_encoder(
    const float* __restrict__ in, int in_dim,
    const float* __restrict__ w0, const float* __restrict__ b0,
    const float* __restrict__ w1, const float* __restrict__ b1,
    const float* __restrict__ w2, const float* __restrict__ b2,
    float* __restrict__ out32, _Float16* __restrict__ out16, int nrows)
{
  __shared__ alignas(16) _Float16 w0s[64 * 32];
  __shared__ alignas(16) _Float16 w1s[64 * 64];
  __shared__ alignas(16) _Float16 w2s[64 * 64];
  __shared__ float b0s[64], b1s[64], b2s[64];
  __shared__ alignas(16) _Float16 stg[8][16 * 64];
  int tid = threadIdx.x;
  for (int i = tid; i < 64 * 32; i += 256) { int n = i >> 5, k = i & 31;
    w0s[i] = (k < in_dim) ? (_Float16)w0[n * in_dim + k] : (_Float16)0.f; }
  for (int i = tid; i < 64 * 64; i += 256) w1s[i] = (_Float16)w1[i];
  for (int i = tid; i < 64 * 64; i += 256) w2s[i] = (_Float16)w2[i];
  if (tid < 64) { b0s[tid] = b0[tid]; b1s[tid] = b1[tid]; b2s[tid] = b2[tid]; }
  __syncthreads();

  int wave = tid >> 5, lane = tid & 31;
  int rbase = blockIdx.x * 128 + wave * 16;
  if (rbase >= nrows) return;
  _Float16* s = stg[wave];
  // stage input rows (K padded to 32 with zeros)
  for (int i = lane; i < 16 * 32; i += 32) { int r = i >> 5, c = i & 31;
    s[r * 64 + c] = (c < in_dim) ? (_Float16)in[(size_t)(rbase + r) * in_dim + c] : (_Float16)0.f; }

  { // layer0: K=32(padded) -> 64, tanh
    v8f acc[4]; int c0 = lane & 15;
#pragma unroll
    for (int t = 0; t < 4; t++) { float bv = b0s[t * 16 + c0];
#pragma unroll
      for (int j = 0; j < 8; j++) acc[t][j] = bv; }
    v16h a0 = ldsA(s, 64, 0, lane);
#pragma unroll
    for (int t = 0; t < 4; t++) acc[t] = wmma_f16(a0, ldsB(w0s, 32, t, 0, lane), acc[t]);
    int rb = (lane >> 4) << 3;
#pragma unroll
    for (int t = 0; t < 4; t++)
#pragma unroll
      for (int j = 0; j < 8; j++)
        s[(rb + j) * 64 + t * 16 + c0] = (_Float16)fast_tanh(acc[t][j]);
  }
  layer64_tanh(s, w1s, b1s, lane);
  { // layer2 + global output
    v8f acc[4]; int c0 = lane & 15;
#pragma unroll
    for (int t = 0; t < 4; t++) { float bv = b2s[t * 16 + c0];
#pragma unroll
      for (int j = 0; j < 8; j++) acc[t][j] = bv; }
    v16h a0 = ldsA(s, 64, 0, lane), a1 = ldsA(s, 64, 32, lane);
#pragma unroll
    for (int t = 0; t < 4; t++) {
      acc[t] = wmma_f16(a0, ldsB(w2s, 64, t, 0, lane), acc[t]);
      acc[t] = wmma_f16(a1, ldsB(w2s, 64, t, 32, lane), acc[t]);
    }
    int rb = (lane >> 4) << 3;
#pragma unroll
    for (int t = 0; t < 4; t++)
#pragma unroll
      for (int j = 0; j < 8; j++) {
        float v = fast_tanh(acc[t][j]);
        size_t g = (size_t)(rbase + rb + j) * 64 + t * 16 + c0;
        if (out32) out32[g] = v; else out16[g] = (_Float16)v;
      }
  }
}

// -------- edge pass 1: msg = relu(x[src]+e)+eps ; atomic max into m[dst] (int-punned, msg>0) --------
__global__ void __launch_bounds__(256) edge_scatter_max(
    const int* __restrict__ ei, const _Float16* __restrict__ e,
    const float* __restrict__ x, int* __restrict__ mmax)
{
  int idx = blockIdx.x * 256 + threadIdx.x;     // exactly N_EDGES*16 threads
  int edge = idx >> 4; int fq = (idx & 15) << 2;
  int src = ei[edge], dst = ei[N_EDGES + edge];
  v4h ev = *(const v4h*)(e + (size_t)edge * 64 + fq);
  float4 xv = *(const float4*)(x + (size_t)src * 64 + fq);
  float m0 = fmaxf(xv.x + (float)ev[0], 0.f) + 1e-7f;
  float m1 = fmaxf(xv.y + (float)ev[1], 0.f) + 1e-7f;
  float m2 = fmaxf(xv.z + (float)ev[2], 0.f) + 1e-7f;
  float m3 = fmaxf(xv.w + (float)ev[3], 0.f) + 1e-7f;
  size_t o = (size_t)dst * 64 + fq;
  atomicMax(mmax + o + 0, __float_as_int(m0));
  atomicMax(mmax + o + 1, __float_as_int(m1));
  atomicMax(mmax + o + 2, __float_as_int(m2));
  atomicMax(mmax + o + 3, __float_as_int(m3));
}

// -------- edge pass 2: ex=exp(msg-m[dst]); den += ex; num += ex*msg --------
__global__ void __launch_bounds__(256) edge_softmax_sum(
    const int* __restrict__ ei, const _Float16* __restrict__ e,
    const float* __restrict__ x, const int* __restrict__ mmax,
    float* __restrict__ den, float* __restrict__ num)
{
  int idx = blockIdx.x * 256 + threadIdx.x;
  int edge = idx >> 4; int fq = (idx & 15) << 2;
  int src = ei[edge], dst = ei[N_EDGES + edge];
  v4h ev = *(const v4h*)(e + (size_t)edge * 64 + fq);
  float4 xv = *(const float4*)(x + (size_t)src * 64 + fq);
  float m0 = fmaxf(xv.x + (float)ev[0], 0.f) + 1e-7f;
  float m1 = fmaxf(xv.y + (float)ev[1], 0.f) + 1e-7f;
  float m2 = fmaxf(xv.z + (float)ev[2], 0.f) + 1e-7f;
  float m3 = fmaxf(xv.w + (float)ev[3], 0.f) + 1e-7f;
  size_t o = (size_t)dst * 64 + fq;
  int4 mb = *(const int4*)(mmax + o);
  float e0 = __expf(m0 - __int_as_float(mb.x));
  float e1 = __expf(m1 - __int_as_float(mb.y));
  float e2 = __expf(m2 - __int_as_float(mb.z));
  float e3 = __expf(m3 - __int_as_float(mb.w));
  atomicAdd(den + o + 0, e0); atomicAdd(num + o + 0, e0 * m0);
  atomicAdd(den + o + 1, e1); atomicAdd(num + o + 1, e1 * m1);
  atomicAdd(den + o + 2, e2); atomicAdd(num + o + 2, e2 * m2);
  atomicAdd(den + o + 3, e3); atomicAdd(num + o + 3, e3 * m3);
}

// -------- conv node update: h=x+num/den; h@w0.T+b0; LN; ReLU; @w1.T+b1 -> x (in place) --------
__global__ void __launch_bounds__(128) conv_node(
    float* __restrict__ x, const float* __restrict__ numr, const float* __restrict__ denr,
    const float* __restrict__ w0, const float* __restrict__ b0,
    const float* __restrict__ lnw, const float* __restrict__ lnb,
    const float* __restrict__ w1, const float* __restrict__ b1, int nrows)
{
  __shared__ alignas(16) _Float16 w0s[128 * 64];
  __shared__ alignas(16) _Float16 w1s[64 * 128];
  __shared__ float b0s[128], lnws[128], lnbs[128], b1s[64];
  __shared__ alignas(16) _Float16 s1[4][16 * 64];
  __shared__ alignas(16) _Float16 s2[4][16 * 128];
  int tid = threadIdx.x;
  for (int i = tid; i < 128 * 64; i += 128) w0s[i] = (_Float16)w0[i];
  for (int i = tid; i < 64 * 128; i += 128) w1s[i] = (_Float16)w1[i];
  if (tid < 128) { b0s[tid] = b0[tid]; lnws[tid] = lnw[tid]; lnbs[tid] = lnb[tid]; }
  if (tid < 64) b1s[tid] = b1[tid];
  __syncthreads();

  int wave = tid >> 5, lane = tid & 31;
  int rbase = blockIdx.x * 64 + wave * 16;
  if (rbase >= nrows) return;
  _Float16* sa = s1[wave];
  _Float16* sb = s2[wave];
  for (int i = lane; i < 16 * 64; i += 32) { int r = i >> 6, c = i & 63;
    size_t g = (size_t)(rbase + r) * 64 + c;
    float d = denr[g];
    float agg = (d > 0.f) ? numr[g] * __builtin_amdgcn_rcpf(d) : 0.f; // empty-dst -> 0
    sa[r * 64 + c] = (_Float16)(x[g] + agg); }

  // GEMM1: 64 -> 128
  v8f acc[8]; int c0 = lane & 15;
#pragma unroll
  for (int t = 0; t < 8; t++) { float bv = b0s[t * 16 + c0];
#pragma unroll
    for (int j = 0; j < 8; j++) acc[t][j] = bv; }
  {
    v16h a0 = ldsA(sa, 64, 0, lane), a1 = ldsA(sa, 64, 32, lane);
#pragma unroll
    for (int t = 0; t < 8; t++) {
      acc[t] = wmma_f16(a0, ldsB(w0s, 64, t, 0, lane), acc[t]);
      acc[t] = wmma_f16(a1, ldsB(w0s, 64, t, 32, lane), acc[t]);
    }
  }
  // LayerNorm over 128 cols, in-register: row sums via xor-shuffles within 16-lane halves
  float sum[8], sq[8];
#pragma unroll
  for (int j = 0; j < 8; j++) { float s_ = 0.f, q_ = 0.f;
#pragma unroll
    for (int t = 0; t < 8; t++) { float v = acc[t][j]; s_ += v; q_ += v * v; }
    sum[j] = s_; sq[j] = q_; }
#pragma unroll
  for (int m_ = 1; m_ < 16; m_ <<= 1)
#pragma unroll
    for (int j = 0; j < 8; j++) {
      sum[j] += __shfl_xor(sum[j], m_, 32);
      sq[j]  += __shfl_xor(sq[j],  m_, 32);
    }
  int rb = (lane >> 4) << 3;
#pragma unroll
  for (int j = 0; j < 8; j++) {
    float mean = sum[j] * (1.f / 128.f);
    float var  = sq[j] * (1.f / 128.f) - mean * mean;
    float rstd = rsqrtf(var + 1e-5f);
#pragma unroll
    for (int t = 0; t < 8; t++) {
      int col = t * 16 + c0;
      float v = (acc[t][j] - mean) * rstd * lnws[col] + lnbs[col];
      v = fmaxf(v, 0.f);
      sb[(rb + j) * 128 + col] = (_Float16)v;
    }
  }
  // GEMM2: 128 -> 64, write back to x
  v8f o[4];
#pragma unroll
  for (int t = 0; t < 4; t++) { float bv = b1s[t * 16 + c0];
#pragma unroll
    for (int j = 0; j < 8; j++) o[t][j] = bv; }
  v16h af[4];
#pragma unroll
  for (int kk = 0; kk < 4; kk++) af[kk] = ldsA(sb, 128, kk * 32, lane);
#pragma unroll
  for (int t = 0; t < 4; t++)
#pragma unroll
    for (int kk = 0; kk < 4; kk++)
      o[t] = wmma_f16(af[kk], ldsB(w1s, 128, t, kk * 32, lane), o[t]);
#pragma unroll
  for (int t = 0; t < 4; t++)
#pragma unroll
    for (int j = 0; j < 8; j++)
      x[(size_t)(rbase + rb + j) * 64 + t * 16 + c0] = o[t][j];
}

// -------- policy head: tanh(64->64), tanh(64->64), dot with pi_w2 row 0; means = 30*(o+b2[0]) --------
__global__ void __launch_bounds__(256) policy_head(
    const float* __restrict__ x,
    const float* __restrict__ w0, const float* __restrict__ b0,
    const float* __restrict__ w1, const float* __restrict__ b1,
    const float* __restrict__ w2, const float* __restrict__ b2,
    float* __restrict__ out, int nrows)
{
  __shared__ alignas(16) _Float16 w0s[64 * 64];
  __shared__ alignas(16) _Float16 w1s[64 * 64];
  __shared__ float b0s[64], b1s[64], w2r0[64];
  __shared__ alignas(16) _Float16 stg[8][16 * 64];
  int tid = threadIdx.x;
  for (int i = tid; i < 64 * 64; i += 256) w0s[i] = (_Float16)w0[i];
  for (int i = tid; i < 64 * 64; i += 256) w1s[i] = (_Float16)w1[i];
  if (tid < 64) { b0s[tid] = b0[tid]; b1s[tid] = b1[tid]; w2r0[tid] = w2[tid]; }
  __syncthreads();

  int wave = tid >> 5, lane = tid & 31;
  int rbase = blockIdx.x * 128 + wave * 16;
  if (rbase >= nrows) return;
  _Float16* s = stg[wave];
  for (int i = lane; i < 16 * 64; i += 32) { int r = i >> 6, c = i & 63;
    s[r * 64 + c] = (_Float16)x[(size_t)(rbase + r) * 64 + c]; }
  layer64_tanh(s, w0s, b0s, lane);
  layer64_tanh(s, w1s, b1s, lane);
  int r = lane & 15, hf = lane >> 4;
  float p = 0.f;
  const _Float16* srow = s + r * 64 + hf * 32;
  const float* wr = w2r0 + hf * 32;
#pragma unroll
  for (int k = 0; k < 32; k++) p += (float)srow[k] * wr[k];
  p += __shfl_xor(p, 16, 32);
  if (hf == 0) out[rbase + r] = 30.f * (p + b2[0]);   // -30 + (o+1)*60/2 == 30*o
}

__global__ void __launch_bounds__(256) zero_f32x4(float* __restrict__ p, int n4) {
  int i = blockIdx.x * 256 + threadIdx.x;
  if (i < n4) ((float4*)p)[i] = make_float4(0.f, 0.f, 0.f, 0.f);
}

extern "C" void kernel_launch(void* const* d_in, const int* in_sizes, int n_in,
                              void* d_out, int out_size, void* d_ws, size_t ws_size,
                              hipStream_t stream) {
  const float* nf    = (const float*)d_in[0];
  const float* ef    = (const float*)d_in[1];
  const int*   ei    = (const int*)d_in[2];
  const float* ne_w0 = (const float*)d_in[3];  const float* ne_b0 = (const float*)d_in[4];
  const float* ne_w1 = (const float*)d_in[5];  const float* ne_b1 = (const float*)d_in[6];
  const float* ne_w2 = (const float*)d_in[7];  const float* ne_b2 = (const float*)d_in[8];
  const float* ee_w0 = (const float*)d_in[9];  const float* ee_b0 = (const float*)d_in[10];
  const float* ee_w1 = (const float*)d_in[11]; const float* ee_b1 = (const float*)d_in[12];
  const float* ee_w2 = (const float*)d_in[13]; const float* ee_b2 = (const float*)d_in[14];
  const float* mp_w0 = (const float*)d_in[15]; const float* mp_b0 = (const float*)d_in[16];
  const float* mp_lnw= (const float*)d_in[17]; const float* mp_lnb= (const float*)d_in[18];
  const float* mp_w1 = (const float*)d_in[19]; const float* mp_b1 = (const float*)d_in[20];
  const float* pi_w0 = (const float*)d_in[21]; const float* pi_b0 = (const float*)d_in[22];
  const float* pi_w1 = (const float*)d_in[23]; const float* pi_b1 = (const float*)d_in[24];
  const float* pi_w2 = (const float*)d_in[25]; const float* pi_b2 = (const float*)d_in[26];
  float* out = (float*)d_out;

  char* ws = (char*)d_ws;
  float*    x   = (float*)(ws);                        // 50000*64*4 = 12.8 MB
  int*      mmx = (int*)(ws + 12800000);               // 12.8 MB (m, den, num contiguous)
  float*    den = (float*)(ws + 25600000);
  float*    num = (float*)(ws + 38400000);
  _Float16* e16 = (_Float16*)(ws + 51200000);          // 800000*64*2 = 102.4 MB

  // encoders
  mlp3_encoder<<<(N_NODES + 127) / 128, 256, 0, stream>>>(
      nf, 11, ne_w0, ne_b0, ne_w1, ne_b1, ne_w2, ne_b2, x, nullptr, N_NODES);
  mlp3_encoder<<<(N_EDGES + 127) / 128, 256, 0, stream>>>(
      ef, 3, ee_w0, ee_b0, ee_w1, ee_b1, ee_w2, ee_b2, nullptr, e16, N_EDGES);

  // 3 GENConv layers
  const int edge_blocks = (N_EDGES * 16) / 256;        // exact: 50000
  for (int i = 0; i < 3; i++) {
    zero_f32x4<<<(2400000 + 255) / 256, 256, 0, stream>>>((float*)mmx, 2400000); // m+den+num
    edge_scatter_max<<<edge_blocks, 256, 0, stream>>>(ei, e16, x, mmx);
    edge_softmax_sum<<<edge_blocks, 256, 0, stream>>>(ei, e16, x, mmx, den, num);
    conv_node<<<(N_NODES + 63) / 64, 128, 0, stream>>>(
        x, num, den,
        mp_w0 + (size_t)i * 128 * 64, mp_b0 + i * 128,
        mp_lnw + i * 128, mp_lnb + i * 128,
        mp_w1 + (size_t)i * 64 * 128, mp_b1 + i * 64, N_NODES);
  }

  // policy head
  policy_head<<<(N_NODES + 127) / 128, 256, 0, stream>>>(
      x, pi_w0, pi_b0, pi_w1, pi_b1, pi_w2, pi_b2, out, N_NODES);
}
